// PhiHarmonicAttention_66254165508719
// MI455X (gfx1250) — matrix operand order repaired and checked by reading, not compiled, not run on hardware
//
#include <hip/hip_runtime.h>
#include <math.h>

typedef _Float16 f16;
typedef __attribute__((ext_vector_type(16))) _Float16 v16h;
typedef __attribute__((ext_vector_type(8)))  float    v8f;
typedef unsigned int u32x4 __attribute__((ext_vector_type(4)));
typedef int          i32x8 __attribute__((ext_vector_type(8)));
typedef int          i32x4 __attribute__((ext_vector_type(4)));

namespace {
constexpr int Bb = 4, S = 1024, E = 1024, Hh = 16, Dd = 64;
constexpr int Kdim = E;             // inner dim of the projection GEMMs
constexpr int Mrows = Bb * S;       // 4096
// TDM-staged A tile: rows of 2048B with a 16B pad inserted every 1024B stored
// (pad_interval=7 -> 1024B, pad_amount=3 -> 16B). Row stride = 2080B = 1040 halfs,
// and columns k>=512 are shifted by +8 halfs. 16B alignment preserved for b128 LDS reads.
constexpr int LDH = 1040;
}

// ---- WMMA fragment loaders (contiguous k order, same for A and B, wave32) ------------
// A (16x32, MxK): lane m=l&15; halfs j=0..15 -> K = kk + hi*16 + j
__device__ inline v16h afrag_f16(const f16* rowk, int hi) {
  v16h a;
  const f16* p0 = rowk + hi * 16;
  const f16* p1 = rowk + hi * 16 + 8;
#pragma unroll
  for (int j = 0; j < 8; ++j) { a[j] = p0[j]; a[j + 8] = p1[j]; }
  return a;
}
__device__ inline v16h afrag_f32(const float* rowk, int hi) {
  v16h a;
  const float* p0 = rowk + hi * 16;
  const float* p1 = rowk + hi * 16 + 8;
#pragma unroll
  for (int j = 0; j < 8; ++j) { a[j] = (f16)p0[j]; a[j + 8] = (f16)p1[j]; }
  return a;
}
// B (32x16, KxN): lane column n=l&15; halfs j -> K = kk + hi*16 + j (contiguous 32B)

__device__ inline v8f wmma_f16(const v16h a, const v16h b, v8f c) {
  return __builtin_amdgcn_wmma_f32_16x16x32_f16(false, a, false, b, (short)0, c,
                                                false, false);
}

__device__ inline float wave_max(float v) {
#pragma unroll
  for (int m = 16; m >= 1; m >>= 1) v = fmaxf(v, __shfl_xor(v, m, 32));
  return v;
}
__device__ inline float wave_sum(float v) {
#pragma unroll
  for (int m = 16; m >= 1; m >>= 1) v += __shfl_xor(v, m, 32);
  return v;
}

// ---- weight prep: W[k][n] f32  ->  WT[n][k] f16 ------------------------------
__global__ __launch_bounds__(256) void convert_wT(const float* __restrict__ W,
                                                  f16* __restrict__ WT) {
  int idx = blockIdx.x * 256 + threadIdx.x;  // idx = n*K + k
  int n = idx >> 10, k = idx & (Kdim - 1);
  WT[idx] = (f16)W[k * E + n];
}

// ---- activation prep: f32 -> f16 elementwise ---------------------------------
__global__ __launch_bounds__(256) void convert_x16(const float* __restrict__ x,
                                                   f16* __restrict__ y) {
  int i = blockIdx.x * 256 + threadIdx.x;
  y[i] = (f16)x[i];
}

// ---- TDM: DMA a [32 x 1024] f16 tile (row stride 1024) into padded LDS -------
__device__ inline void tdm_stage_tile(const f16* gsrc, f16* lds_dst) {
  u32x4 g0;
  unsigned long long ga = (unsigned long long)(size_t)gsrc;
  g0[0] = 1u;                                   // count=1, user descriptor
  g0[1] = (unsigned)(size_t)lds_dst;            // LDS byte offset (addr[31:0])
  g0[2] = (unsigned)ga;                         // global_addr[31:0]
  g0[3] = (unsigned)(ga >> 32) | (2u << 30);    // global_addr[56:32] | type=2
  i32x8 g1;
  g1[0] = (1 << 16)        // data_size = 1 (2 bytes)
        | (1 << 20)        // pad_enable
        | (7 << 22)        // pad_interval = 1024B
        | (3 << 25);       // pad_amount   = 16B
  g1[1] = (int)(1024u << 16);   // tensor_dim0 = 1024 (low 16 bits)
  g1[2] = (int)((unsigned)Mrows << 16);  // tensor_dim1 low 16 bits
  g1[3] = (int)(1024u << 16);   // tile_dim0 = 1024
  g1[4] = 32;                   // tile_dim1 = 32 (tile_dim2 = 0 -> 2D)
  g1[5] = 1024;                 // tensor_dim0_stride
  g1[6] = 0;
  g1[7] = 0;
  i32x4 z4 = {0, 0, 0, 0};          // groups 2/3 unused for 2D tiles
  i32x8 z8 = {0, 0, 0, 0, 0, 0, 0, 0};
  __builtin_amdgcn_tensor_load_to_lds(g0, g1, z4, z4, z8, 0);
  __builtin_amdgcn_s_wait_tensorcnt(0);
}

// ---- generic 4096x1024x1024 GEMM, f16 WMMA, fp32 accumulate ------------------
// Block tile 32(M) x 128(N); each wave owns a 2x2 register tile (32x32):
// 4 WMMAs per {2 A-frag LDS loads + 2 B-frag global loads}.
// mode 0: out f16 [B,H,S,D] (Q,K)   mode 1: out f16 [B,H,D,S] (V transposed)
// mode 2: out f32 row-major [M,E]   (final projection -> d_out)
__device__ inline void store_acc(const v8f acc, int mbase, int nl, int hi,
                                 const float* bias, void* out, int mode) {
  const float bsv = bias[nl];
  if (mode == 2) {
    float* o = (float*)out;
#pragma unroll
    for (int r = 0; r < 8; ++r) {
      int m = mbase + r + hi * 8;
      o[(size_t)m * E + nl] = acc[r] + bsv;
    }
  } else {
    f16* o = (f16*)out;
    const int h = nl >> 6, d = nl & 63;
#pragma unroll
    for (int r = 0; r < 8; ++r) {
      int m = mbase + r + hi * 8;
      int b = m >> 10, s = m & (S - 1);
      size_t idx = (mode == 0) ? ((((size_t)b * Hh + h) * S + s) * Dd + d)
                               : ((((size_t)b * Hh + h) * Dd + d) * S + s);
      o[idx] = (f16)(acc[r] + bsv);
    }
  }
}

__global__ __launch_bounds__(128) void gemm_wmma(const f16* __restrict__ X16,
                                                 const f16* __restrict__ WT,
                                                 const float* __restrict__ bias,
                                                 void* __restrict__ out, int mode) {
  __shared__ __align__(32) f16 At[32 * LDH];    // ~65 KB of the 320 KB WGP LDS
  const int tid = threadIdx.x, lane = tid & 31, wave = tid >> 5;
  const int m0 = blockIdx.x * 32;
  const int n0 = blockIdx.y * 128 + wave * 32;

  if (wave == 0) tdm_stage_tile(X16 + (size_t)m0 * Kdim, At);
  __syncthreads();

  const int ml = lane & 15, hi = lane >> 4;
  const f16* w0 = WT + (size_t)(n0 + ml) * Kdim;
  const f16* w1 = WT + (size_t)(n0 + 16 + ml) * Kdim;
  const f16* ar0 = At + ml * LDH;
  const f16* ar1 = At + (16 + ml) * LDH;
  v8f acc00 = {}, acc01 = {}, acc10 = {}, acc11 = {};
  for (int kk = 0; kk < Kdim; kk += 32) {
    const int ko = kk + (kk >= 512 ? 8 : 0);    // skip the TDM mid-row pad
    v16h a0 = afrag_f16(ar0 + ko, hi);
    v16h a1 = afrag_f16(ar1 + ko, hi);
    v16h b0 = *(const v16h*)(w0 + kk + hi * 16);
    v16h b1 = *(const v16h*)(w1 + kk + hi * 16);
    acc00 = wmma_f16(a0, b0, acc00);
    acc01 = wmma_f16(a0, b1, acc01);
    acc10 = wmma_f16(a1, b0, acc10);
    acc11 = wmma_f16(a1, b1, acc11);
  }
  store_acc(acc00, m0,      n0 + ml,      hi, bias, out, mode);
  store_acc(acc01, m0,      n0 + 16 + ml, hi, bias, out, mode);
  store_acc(acc10, m0 + 16, n0 + ml,      hi, bias, out, mode);
  store_acc(acc11, m0 + 16, n0 + 16 + ml, hi, bias, out, mode);
}

// ---- cw = softmax(query@Wc + bc); mult[b,h,s] = 1 + cons[b,s]*cw -------------
__global__ __launch_bounds__(128) void cw_mult_kernel(const float* __restrict__ query,
                                                      const float* __restrict__ Wc,
                                                      const float* __restrict__ bc,
                                                      const float* __restrict__ cons,
                                                      float* __restrict__ mult) {
  int bs = blockIdx.x;                 // b*S + s
  int b = bs >> 10, s = bs & (S - 1);
  int tid = threadIdx.x;
  __shared__ float qrow[E];
  __shared__ float logit[Hh];
  for (int i = tid; i < E; i += 128) qrow[i] = query[(size_t)bs * E + i];
  __syncthreads();
  int h = tid >> 3, part = tid & 7;    // 16 heads x 8 collaborating lanes
  float acc = 0.f;
  for (int k = part; k < E; k += 8) acc += qrow[k] * Wc[k * Hh + h];
  acc += __shfl_xor(acc, 1, 32);
  acc += __shfl_xor(acc, 2, 32);
  acc += __shfl_xor(acc, 4, 32);
  if (part == 0) logit[h] = acc + bc[h];
  __syncthreads();
  if (tid == 0) {
    float mx = -__builtin_inff();
    for (int i = 0; i < Hh; ++i) mx = fmaxf(mx, logit[i]);
    float den = 0.f;
    for (int i = 0; i < Hh; ++i) den += __expf(logit[i] - mx);
    float cv = cons[bs];
    for (int i = 0; i < Hh; ++i) {
      float p = __expf(logit[i] - mx) / den;
      mult[((size_t)b * Hh + i) * S + s] = 1.f + cv * p;
    }
  }
}

// ---- fused attention: scores -> softmax1 -> softmax2(c*p, unmasked) -> attn@V
// one block per (b,h,16-query tile); 16 waves; attn written to HBM exactly once
__global__ __launch_bounds__(512) void attn_fused(const f16* __restrict__ q16,
                                                  const f16* __restrict__ k16,
                                                  const f16* __restrict__ vT16,
                                                  const float* __restrict__ mult,
                                                  const float* __restrict__ phi_bias,
                                                  float* __restrict__ attn_out,
                                                  f16* __restrict__ ctx16,
                                                  float inv_scale) {
  constexpr int LD = S + 4;            // padded f32 row stride (bank stride 4)
  __shared__ __align__(32) float sc[16 * LD];   // ~64.3 KB
  const int qt = blockIdx.x, h = blockIdx.y, b = blockIdx.z;
  const int tid = threadIdx.x, lane = tid & 31, wave = tid >> 5;
  const int m = lane & 15, hi = lane >> 4;
  const size_t bh = (size_t)b * Hh + h;

  // Q tile A-fragments (D=64 -> two K=32 chunks), held in registers
  const f16* qrow = q16 + (bh * S + qt * 16 + m) * Dd;
  const v16h a0 = afrag_f16(qrow + 0, hi);
  const v16h a1 = afrag_f16(qrow + 32, hi);
  const float pb = phi_bias[h];

  // phase 1: score tiles for kt <= qt (causal); mask inside diagonal tile
  const f16* kbase = k16 + bh * S * Dd;
  for (int kt = wave; kt <= qt; kt += 16) {
    const f16* krow = kbase + (size_t)(kt * 16 + m) * Dd;  // lane column = key
    v16h b0 = *(const v16h*)(krow + hi * 16);
    v16h b1 = *(const v16h*)(krow + 32 + hi * 16);
    v8f c = {};
    c = wmma_f16(a0, b0, c);
    c = wmma_f16(a1, b1, c);
    int kg = kt * 16 + m;
#pragma unroll
    for (int r = 0; r < 8; ++r) {
      int qr = r + hi * 8;
      int qg = qt * 16 + qr;
      float v = c[r] * inv_scale + pb;
      if (kg > qg) v = -__builtin_inff();
      sc[qr * LD + kg] = v;
    }
  }
  __syncthreads();

  // phase 2: per-row double softmax (wave r owns row r)
  {
    const int row = wave;
    const int qg = qt * 16 + row;
    const int kend = (qt + 1) * 16;
    float* srow = sc + row * LD;
    float mx = -__builtin_inff();
    for (int k = lane; k < kend; k += 32) mx = fmaxf(mx, srow[k]);
    mx = wave_max(mx);
    float den = 0.f;
    for (int k = lane; k < kend; k += 32) den += __expf(srow[k] - mx);
    den = wave_sum(den);
    const float inv_den = 1.f / den;
    const float cm = mult[bh * S + qg];
    const float m2 = cm * inv_den;          // max of c*p; masked entries give 0 <= m2
    const float e_tail = __expf(-m2);       // exp(0 - m2) for masked entries
    float z2 = 0.f;
    for (int k = lane; k < kend; k += 32) {
      float p = __expf(srow[k] - mx) * inv_den;   // masked -> 0
      float e = __expf(cm * p - m2);
      srow[k] = e;
      z2 += e;
    }
    z2 = wave_sum(z2) + (float)(S - kend) * e_tail;
    const float invz = 1.f / z2;
    const float tail = e_tail * invz;
    float* arow = attn_out + (bh * S + qg) * (size_t)S;
    for (int k = lane; k < kend; k += 32) {
      float v = srow[k] * invz;
      srow[k] = v;                 // final attn kept in LDS for attn@V
      arow[k] = v;
    }
    for (int k = kend + lane; k < S; k += 32) {
      srow[k] = tail;              // unmasked softmax2 leaks uniform tail
      arow[k] = tail;
    }
  }
  __syncthreads();

  // phase 3: ctx[16q x 64d] = attn[16x1024] @ V ; waves 0..3 take d-subtiles
  if (wave < 4) {
    const int d = wave * 16 + m;
    const f16* vrow = vT16 + (bh * Dd + d) * (size_t)S;   // V transposed [B,H,D,S]
    const float* arow = sc + m * LD;                      // lane m = attn row
    v8f c = {};
    for (int kk = 0; kk < S; kk += 32) {
      v16h a = afrag_f32(arow + kk, hi);
      v16h bf = *(const v16h*)(vrow + kk + hi * 16);
      c = wmma_f16(a, bf, c);
    }
#pragma unroll
    for (int r = 0; r < 8; ++r) {
      int qg = qt * 16 + r + hi * 8;
      ctx16[((size_t)b * S + qg) * E + h * Dd + d] = (f16)c[r];
    }
  }
}

extern "C" void kernel_launch(void* const* d_in, const int* in_sizes, int n_in,
                              void* d_out, int out_size, void* d_ws, size_t ws_size,
                              hipStream_t stream) {
  const float* query = (const float*)d_in[0];
  const float* key_  = (const float*)d_in[1];
  const float* value = (const float*)d_in[2];
  const float* cons  = (const float*)d_in[3];
  // d_in[4] attn_mask: causal tril, applied analytically — unused
  const float* Wq = (const float*)d_in[5];
  const float* bq = (const float*)d_in[6];
  const float* Wk = (const float*)d_in[7];
  const float* bk = (const float*)d_in[8];
  const float* Wv = (const float*)d_in[9];
  const float* bv = (const float*)d_in[10];
  const float* Wo = (const float*)d_in[11];
  const float* bo = (const float*)d_in[12];
  const float* Wc = (const float*)d_in[13];
  const float* bc = (const float*)d_in[14];
  const float* phi_bias = (const float*)d_in[15];

  char* p = (char*)d_ws;
  f16* wqT = (f16*)p; p += (size_t)E * E * 2;
  f16* wkT = (f16*)p; p += (size_t)E * E * 2;
  f16* wvT = (f16*)p; p += (size_t)E * E * 2;
  f16* woT = (f16*)p; p += (size_t)E * E * 2;
  f16* q16 = (f16*)p; p += (size_t)Bb * Hh * S * Dd * 2;
  f16* k16 = (f16*)p; p += (size_t)Bb * Hh * S * Dd * 2;
  f16* vT16 = (f16*)p; p += (size_t)Bb * Hh * S * Dd * 2;
  f16* qx16 = (f16*)p; p += (size_t)Mrows * E * 2;
  f16* kx16 = (f16*)p; p += (size_t)Mrows * E * 2;
  f16* vx16 = (f16*)p; p += (size_t)Mrows * E * 2;
  f16* ctx16 = (f16*)p; p += (size_t)Mrows * E * 2;
  float* mult = (float*)p; p += (size_t)Bb * Hh * S * 4;

  float* out_ = (float*)d_out;
  float* attn_out = out_ + (size_t)Bb * S * E;

  convert_wT<<<E * E / 256, 256, 0, stream>>>(Wq, wqT);
  convert_wT<<<E * E / 256, 256, 0, stream>>>(Wk, wkT);
  convert_wT<<<E * E / 256, 256, 0, stream>>>(Wv, wvT);
  convert_wT<<<E * E / 256, 256, 0, stream>>>(Wo, woT);

  const int nx = Mrows * E;
  convert_x16<<<nx / 256, 256, 0, stream>>>(query, qx16);
  convert_x16<<<nx / 256, 256, 0, stream>>>(key_,  kx16);
  convert_x16<<<nx / 256, 256, 0, stream>>>(value, vx16);

  dim3 gg(Mrows / 32, E / 128);
  gemm_wmma<<<gg, 128, 0, stream>>>(qx16, wqT, bq, q16, 0);
  gemm_wmma<<<gg, 128, 0, stream>>>(kx16, wkT, bk, k16, 0);
  gemm_wmma<<<gg, 128, 0, stream>>>(vx16, wvT, bv, vT16, 1);

  cw_mult_kernel<<<Bb * S, 128, 0, stream>>>(query, Wc, bc, cons, mult);

  const float PHIc = 1.61803398874989484820f;
  const float inv_scale = 1.0f / (sqrtf((float)Dd) * sqrtf(PHIc));
  attn_fused<<<dim3(S / 16, Hh, Bb), 512, 0, stream>>>(q16, k16, vT16, mult, phi_bias,
                                                       attn_out, ctx16, inv_scale);

  gemm_wmma<<<gg, 128, 0, stream>>>(ctx16, woT, bo, out_, 2);
}
